// GAT_38981123178605
// MI455X (gfx1250) — matrix operand order repaired and checked by reading, not compiled, not run on hardware
//
#include <hip/hip_runtime.h>
#include <hip/hip_bf16.h>
#include <math.h>

typedef __attribute__((ext_vector_type(2)))  float  v2f;
typedef __attribute__((ext_vector_type(8)))  float  v8f;
typedef __bf16 bf16_t;
typedef __attribute__((ext_vector_type(16))) __bf16 v16bf;

#define GAT_N 8
#define GAT_L 2048
#define GAT_D 128
#define ALPHA 0.2f
#define NEG_INF (-9e15f)

// ---------------------------------------------------------------------------
// K1: h = x @ W (fp32, V_WMMA_F32_16X16X4_F32). Also writes hbT, a TRANSPOSED
// bf16 copy (hbT[n][d][m]) so K4 can feed the WMMA B operand with b128 loads
// of 16 contiguous K-values. grid = 1024, block = 256 (8 waves = 8 col tiles)
// ---------------------------------------------------------------------------
__global__ void __launch_bounds__(256)
gat_proj(const float* __restrict__ x, const float* __restrict__ W,
         float* __restrict__ h, bf16_t* __restrict__ hbT) {
    const int lane = threadIdx.x & 31;
    const int wave = threadIdx.x >> 5;        // col tile 0..7
    const int rowTile = blockIdx.x;           // 0..1023
    const int r  = lane & 15;
    const int hi = lane >> 4;                 // 0/1 -> K offset 0/2
    const int col = wave * 16 + r;
    const float* xrow = x + ((long)rowTile * 16 + r) * GAT_D;

    v8f c = {};
    for (int k = 0; k < GAT_D; k += 4) {
        v2f a, b;
        a.x = xrow[k + 2 * hi];
        a.y = xrow[k + 2 * hi + 1];
        b.x = W[(k + 2 * hi) * GAT_D + col];
        b.y = W[(k + 2 * hi + 1) * GAT_D + col];
        c = __builtin_amdgcn_wmma_f32_16x16x4_f32(false, a, false, b,
                                                  (short)0, c, false, false);
    }
    #pragma unroll
    for (int j = 0; j < 8; ++j) {
        long R = (long)rowTile * 16 + j + 8 * hi;     // D-layout: M=j / M=8+j
        h[R * GAT_D + col] = c[j];
        long n = R >> 11, l = R & (GAT_L - 1);
        hbT[(n * GAT_D + col) * GAT_L + l] = (bf16_t)c[j];  // transposed bf16
    }
}

// ---------------------------------------------------------------------------
// K2: f1 = h.a1, f2 = h.a2 per row. Wave per row, butterfly reduce (wave32).
// ---------------------------------------------------------------------------
__global__ void __launch_bounds__(256)
gat_attvec(const float* __restrict__ h, const float* __restrict__ a1,
           const float* __restrict__ a2, float* __restrict__ f1,
           float* __restrict__ f2) {
    const int lane = threadIdx.x & 31;
    const int wave = threadIdx.x >> 5;
    const long row = (long)blockIdx.x * 8 + wave;
    const float* hr = h + row * GAT_D;
    float s1 = 0.f, s2 = 0.f;
    #pragma unroll
    for (int i = 0; i < GAT_D / 32; ++i) {
        float hv = hr[lane + 32 * i];
        s1 += hv * a1[lane + 32 * i];
        s2 += hv * a2[lane + 32 * i];
    }
    #pragma unroll
    for (int off = 16; off > 0; off >>= 1) {
        s1 += __shfl_xor(s1, off, 32);
        s2 += __shfl_xor(s2, off, 32);
    }
    if (lane == 0) { f1[row] = s1; f2[row] = s2; }
}

// ---------------------------------------------------------------------------
// K3: per-row softmax stats (online max/sum) over masked lrelu scores, fp32.
// b128-vectorized first pass over adj (134 MB HBM -> resident in 192 MB L2).
// ---------------------------------------------------------------------------
__global__ void __launch_bounds__(256)
gat_rowstat(const float* __restrict__ f1, const float* __restrict__ f2,
            const int* __restrict__ adj, float* __restrict__ rowmax,
            float* __restrict__ rowinv) {
    const int lane = threadIdx.x & 31;
    const int wave = threadIdx.x >> 5;
    const long R = (long)blockIdx.x * 8 + wave;       // global row 0..16383
    const int  n = (int)(R >> 11);                    // R / 2048
    const float f1v = f1[R];
    const float* f2n = f2 + (long)n * GAT_L;
    const int* adjr = adj + R * GAT_L;

    float mrun = -INFINITY, srun = 0.f;
    for (int mb = 0; mb < GAT_L; mb += 128) {
        const int m0 = mb + lane * 4;
        int4   ad = *(const int4*)(adjr + m0);
        float4 g  = *(const float4*)(f2n + m0);
        const float gv[4] = {g.x, g.y, g.z, g.w};
        const int   av[4] = {ad.x, ad.y, ad.z, ad.w};
        #pragma unroll
        for (int e = 0; e < 4; ++e) {
            float ev = f1v + gv[e];
            ev = ev > 0.f ? ev : ALPHA * ev;
            float v = av[e] > 0 ? ev : NEG_INF;
            if (v > mrun) {
                srun = srun * __expf(mrun - v) + 1.f; // exp(-inf)=0 first time
                mrun = v;
            } else {
                srun += __expf(v - mrun);
            }
        }
    }
    #pragma unroll
    for (int off = 16; off > 0; off >>= 1) {
        float mo = __shfl_xor(mrun, off, 32);
        float so = __shfl_xor(srun, off, 32);
        float mn = fmaxf(mrun, mo);
        srun = srun * __expf(mrun - mn) + so * __expf(mo - mn);
        mrun = mn;
    }
    if (lane == 0) { rowmax[R] = mrun; rowinv[R] = 1.0f / srun; }
}

// ---------------------------------------------------------------------------
// K4: out = elu(softmax(scores) @ h), PV product via V_WMMA_F32_16X16X32_BF16
// (K=32 per issue, fp32 accumulate). Workgroup per (n, 16-row tile); 4 waves
// x 32 output columns. A (probabilities) generated per-lane in 16-bit
// A-layout from b128 adj/f2 reads; B fed by b128 loads of transposed bf16 h
// (16 contiguous K per lane -> same K order as A). adj hits L2.
// grid = 1024, block = 128.
// ---------------------------------------------------------------------------
__global__ void __launch_bounds__(128)
gat_out(const bf16_t* __restrict__ hbT, const float* __restrict__ f1,
        const float* __restrict__ f2, const int* __restrict__ adj,
        const float* __restrict__ rowmax, const float* __restrict__ rowinv,
        float* __restrict__ out) {
    const int lane = threadIdx.x & 31;
    const int wave = threadIdx.x >> 5;        // 0..3
    const int n     = blockIdx.x >> 7;        // /128
    const int lTile = blockIdx.x & 127;
    const int r  = lane & 15;                 // A/B/D row-within-tile
    const int hi = lane >> 4;
    const long R = (long)n * GAT_L + lTile * 16 + r;   // this lane's query row

    const float f1v = f1[R];
    const float mx  = rowmax[R];
    const float inv = rowinv[R];
    const float*  f2n  = f2  + (long)n * GAT_L;
    const int*    adjr = adj + R * (long)GAT_L;
    const bf16_t* hTn  = hbT + (long)n * GAT_D * GAT_L;
    const int col0 = wave * 32 + r;                    // first d-tile column
    const bf16_t* bp0 = hTn + (long)col0 * GAT_L;      // column streams
    const bf16_t* bp1 = hTn + (long)(col0 + 16) * GAT_L;

    v8f acc0 = {};
    v8f acc1 = {};
    #pragma unroll 2
    for (int mb = 0; mb < GAT_L; mb += 32) {
        __builtin_prefetch(adjr + mb + 128, 0, 0);    // global_prefetch_b8
        // ---- A tile: 16 probabilities per lane in bf16 A-layout.
        // lane<16: K = e (e=0..15); lane>=16: +16.
        const int mA0 = mb + 16 * hi;         // e = 0..7  -> m = mA0 + e
        const int mA1 = mb + 16 * hi + 8;     // e = 8..15 -> m = mA1 + (e-8)
        int4 ad0 = *(const int4*)(adjr + mA0);
        int4 ad1 = *(const int4*)(adjr + mA0 + 4);
        int4 ad2 = *(const int4*)(adjr + mA1);
        int4 ad3 = *(const int4*)(adjr + mA1 + 4);
        float4 g0 = *(const float4*)(f2n + mA0);
        float4 g1 = *(const float4*)(f2n + mA0 + 4);
        float4 g2 = *(const float4*)(f2n + mA1);
        float4 g3 = *(const float4*)(f2n + mA1 + 4);
        const float fv[16] = {g0.x, g0.y, g0.z, g0.w, g1.x, g1.y, g1.z, g1.w,
                              g2.x, g2.y, g2.z, g2.w, g3.x, g3.y, g3.z, g3.w};
        const int   av[16] = {ad0.x, ad0.y, ad0.z, ad0.w, ad1.x, ad1.y, ad1.z, ad1.w,
                              ad2.x, ad2.y, ad2.z, ad2.w, ad3.x, ad3.y, ad3.z, ad3.w};
        union { v16bf v; unsigned u[8]; } A;
        #pragma unroll
        for (int e = 0; e < 16; ++e) {
            float ev = f1v + fv[e];
            ev = ev > 0.f ? ev : ALPHA * ev;
            float val = av[e] > 0 ? ev : NEG_INF;
            float p = __expf(val - mx) * inv;         // fully-masked row -> 1/cnt
            A.v[e] = (bf16_t)p;
        }
        // ---- B tiles: lane holds K = kb..kb+15 at fixed column -> contiguous
        // 32 B in hbT; dword j = {K=kb+2j (lo), kb+2j+1 (hi)} (same K order as A).
        const int kb = mb + 16 * hi;          // lanes>=16 hold K=16..31
        union { v16bf v; uint4 q[2]; } B0, B1;
        B0.q[0] = *(const uint4*)(bp0 + kb);
        B0.q[1] = *(const uint4*)(bp0 + kb + 8);
        B1.q[0] = *(const uint4*)(bp1 + kb);
        B1.q[1] = *(const uint4*)(bp1 + kb + 8);
        acc0 = __builtin_amdgcn_wmma_f32_16x16x32_bf16(false, A.v, false, B0.v,
                                                       (short)0, acc0, false, false);
        acc1 = __builtin_amdgcn_wmma_f32_16x16x32_bf16(false, A.v, false, B1.v,
                                                       (short)0, acc1, false, false);
    }
    #pragma unroll
    for (int j = 0; j < 8; ++j) {
        long orow = (long)n * GAT_L + lTile * 16 + j + 8 * hi;
        float v0 = acc0[j];
        float v1 = acc1[j];
        out[orow * GAT_D + wave * 32 + r]      = v0 > 0.f ? v0 : __expf(v0) - 1.f;
        out[orow * GAT_D + wave * 32 + 16 + r] = v1 > 0.f ? v1 : __expf(v1) - 1.f;
    }
}

// ---------------------------------------------------------------------------
extern "C" void kernel_launch(void* const* d_in, const int* in_sizes, int n_in,
                              void* d_out, int out_size, void* d_ws, size_t ws_size,
                              hipStream_t stream) {
    const float* x   = (const float*)d_in[0];   // [8,2048,128]
    const int*   adj = (const int*)  d_in[1];   // [8,2048,2048]
    const float* W   = (const float*)d_in[2];   // [128,128]
    const float* a1  = (const float*)d_in[3];   // [128]
    const float* a2  = (const float*)d_in[4];   // [128]
    float* out = (float*)d_out;                 // [8,2048,128]

    char* ws = (char*)d_ws;
    const long ROWS = (long)GAT_N * GAT_L;      // 16384
    float*  h    = (float*)ws;                                   // 8 MB fp32
    bf16_t* hbT  = (bf16_t*)(ws + ROWS * GAT_D * sizeof(float)); // 4 MB bf16 (transposed)
    float*  f1   = (float*)(ws + ROWS * GAT_D * (sizeof(float) + sizeof(bf16_t)));
    float*  f2   = f1 + ROWS;
    float*  rmax = f2 + ROWS;
    float*  rinv = rmax + ROWS;

    gat_proj   <<<ROWS / 16, 256, 0, stream>>>(x, W, h, hbT);
    gat_attvec <<<ROWS / 8,  256, 0, stream>>>(h, a1, a2, f1, f2);
    gat_rowstat<<<ROWS / 8,  256, 0, stream>>>(f1, f2, adj, rmax, rinv);
    gat_out    <<<ROWS / 16, 128, 0, stream>>>(hbT, f1, f2, adj, rmax, rinv, out);
}
